// GAEWithAttributes_54743653154845
// MI455X (gfx1250) — matrix operand order repaired and checked by reading, not compiled, not run on hardware
//
#include <hip/hip_runtime.h>
#include <hip/hip_bf16.h>
#include <math.h>

typedef __attribute__((ext_vector_type(16))) _Float16 v16h;
typedef __attribute__((ext_vector_type(8)))  float    v8f;

#define NEG_SLOPE 0.2f

// ---------------------------------------------------------------------------
// WMMA GEMM: C[M,N] = A[M,KT] @ B[KT,N] + bias (optional ReLU)
// KT compile-time (64/128), KT % 32 == 0, A 16-byte aligned, KT*N <= 8192.
// B staged in LDS as f16 in WMMA B-fragment order:
//   LDS index for B[k][col] = ((k>>5)*N + col)*32 + (k&31)
// -> lane (l,half)'s 16 halves for chunk c, col are one contiguous v16h at
//    (c*N + col)*32 + half*16 (32B aligned -> 2x ds_load_b128).
// Per chunk: load ALL NT B fragments, then issue NT WMMAs back-to-back
// (single s_wait_dscnt per chunk). Chunk loop fully unrolled (KT constant).
// M/N tails via index clamping (EXEC all-1s, required by WMMA) + store checks.
// 4 waves/block, each wave a 16-row strip. A/B k = 16*half + s (contiguous); acc per CDNA5 ISA 7.12.2.
// ---------------------------------------------------------------------------
template<int NT, int KT, bool RELU>
__global__ __launch_bounds__(128)
void wmma_gemm_kernel(const float* __restrict__ A, const float* __restrict__ B,
                      const float* __restrict__ bias, float* __restrict__ C,
                      int M, int N) {
    __shared__ _Float16 Bs[8192];     // 16 KB, fits KT*N for all call sites

    const int lane = threadIdx.x & 31;
    const int wave = threadIdx.x >> 5;
    const int half = lane >> 4;       // 0 or 1
    const int l    = lane & 15;

    // ---- cooperative B convert+pack into LDS (all waves participate) ----
    for (int k = wave; k < KT; k += 4) {
        const int base = ((k >> 5) * N) * 32 + (k & 31);
        for (int col = lane; col < N; col += 32)
            Bs[base + col * 32] = (_Float16)B[(size_t)k * N + col];
    }
    __syncthreads();

    const int tileM = (blockIdx.x * 4 + wave) * 16;
    if (tileM >= M) return;           // wave-uniform exit (after barrier)

    int rowA = tileM + l;
    if (rowA >= M) rowA = M - 1;      // clamp: redundant load, discarded at store
    const float* __restrict__ Arow = A + (size_t)rowA * KT;

    v8f acc[NT] = {};

#pragma unroll
    for (int k0 = 0; k0 < KT; k0 += 32) {
        // ---- A fragment: two contiguous 8-float runs -> 4x b128 loads ----
        v16h a;
        {
            const float4* p0 = (const float4*)(Arow + k0 + half * 16);
            const float4* p1 = (const float4*)(Arow + k0 + half * 16 + 8);
            float4 u0 = p0[0], u1 = p0[1], u2 = p1[0], u3 = p1[1];
            a[0]  = (_Float16)u0.x; a[1]  = (_Float16)u0.y;
            a[2]  = (_Float16)u0.z; a[3]  = (_Float16)u0.w;
            a[4]  = (_Float16)u1.x; a[5]  = (_Float16)u1.y;
            a[6]  = (_Float16)u1.z; a[7]  = (_Float16)u1.w;
            a[8]  = (_Float16)u2.x; a[9]  = (_Float16)u2.y;
            a[10] = (_Float16)u2.z; a[11] = (_Float16)u2.w;
            a[12] = (_Float16)u3.x; a[13] = (_Float16)u3.y;
            a[14] = (_Float16)u3.z; a[15] = (_Float16)u3.w;
        }
        const int cbase = (k0 >> 5) * N;

        // ---- load all NT B fragments first (clause + one dscnt wait) ----
        v16h bf[NT];
#pragma unroll
        for (int t = 0; t < NT; ++t) {
            int col = t * 16 + l;
            if (col >= N) col = N - 1;       // clamp (N-tail, NT=1 case only)
            bf[t] = *(const v16h*)&Bs[(cbase + col) * 32 + half * 16];
        }
        // ---- then stream the WMMAs back-to-back ----
#pragma unroll
        for (int t = 0; t < NT; ++t)
            acc[t] = __builtin_amdgcn_wmma_f32_16x16x32_f16(
                false, a, false, bf[t], (short)0, acc[t], false, false);
    }

#pragma unroll
    for (int t = 0; t < NT; ++t) {
        int col = t * 16 + l;
        if (col < N) {
            float bv = bias ? bias[col] : 0.0f;
#pragma unroll
            for (int r = 0; r < 8; ++r) {
                int m = tileM + r + half * 8;
                if (m < M) {
                    float v = acc[t][r] + bv;
                    if (RELU) v = fmaxf(v, 0.0f);
                    C[(size_t)m * N + col] = v;
                }
            }
        }
    }
}

// ---------------------------------------------------------------------------
// Elementwise / edge kernels
// ---------------------------------------------------------------------------
__global__ void k_fill(float* __restrict__ p, float v, long n) {
    long i = (long)blockIdx.x * blockDim.x + threadIdx.x;
    if (i < n) p[i] = v;
}

__global__ void k_relu(float* __restrict__ p, long n) {
    long i = (long)blockIdx.x * blockDim.x + threadIdx.x;
    if (i < n) p[i] = fmaxf(p[i], 0.0f);
}

// score_src[n] = h[n,:]·a_src ; score_dst[n] = h[n,:]·a_dst   (F = 64)
__global__ void k_scores(const float* __restrict__ h,
                         const float* __restrict__ asrc,
                         const float* __restrict__ adst,
                         float* __restrict__ ss, float* __restrict__ sd, int Nn) {
    int n = blockIdx.x * blockDim.x + threadIdx.x;
    if (n >= Nn) return;
    float s0 = 0.0f, s1 = 0.0f;
    const float* hp = h + (size_t)n * 64;
#pragma unroll 4
    for (int f = 0; f < 64; f += 4) {
        float4 hv = *(const float4*)(hp + f);
        float4 as = *(const float4*)(asrc + f);
        float4 ad = *(const float4*)(adst + f);
        s0 += hv.x * as.x + hv.y * as.y + hv.z * as.z + hv.w * as.w;
        s1 += hv.x * ad.x + hv.y * ad.y + hv.z * ad.z + hv.w * ad.w;
    }
    ss[n] = s0;
    sd[n] = s1;
}

// alpha[e] = leaky_relu(score_src[s] + score_dst[d]); edges then self-loops
__global__ void k_alpha(const float* __restrict__ ss, const float* __restrict__ sd,
                        const int* __restrict__ src, const int* __restrict__ dst,
                        float* __restrict__ alpha, int E, int Nn) {
    int i = blockIdx.x * blockDim.x + threadIdx.x;
    int T = E + Nn;
    if (i >= T) return;
    int s = (i < E) ? src[i] : (i - E);
    int d = (i < E) ? dst[i] : (i - E);
    float v = ss[s] + sd[d];
    alpha[i] = (v > 0.0f) ? v : NEG_SLOPE * v;
}

__device__ inline void atomicMaxF(float* addr, float value) {
    if (value >= 0.0f)
        atomicMax((int*)addr, __float_as_int(value));
    else
        atomicMin((unsigned int*)addr, __float_as_uint(value));
}

__global__ void k_segmax(const float* __restrict__ alpha, const int* __restrict__ dst,
                         float* __restrict__ amax, int E, int Nn) {
    int i = blockIdx.x * blockDim.x + threadIdx.x;
    int T = E + Nn;
    if (i >= T) return;
    int d = (i < E) ? dst[i] : (i - E);
    atomicMaxF(&amax[d], alpha[i]);
}

// alpha <- exp(alpha - amax[d]); denom[d] += alpha
__global__ void k_expsum(float* __restrict__ alpha, const float* __restrict__ amax,
                         const int* __restrict__ dst, float* __restrict__ denom,
                         int E, int Nn) {
    int i = blockIdx.x * blockDim.x + threadIdx.x;
    int T = E + Nn;
    if (i >= T) return;
    int d = (i < E) ? dst[i] : (i - E);
    float e = __expf(alpha[i] - amax[d]);
    alpha[i] = e;
    atomicAdd(&denom[d], e);
}

// out[n,:] = bias  (scatter target init)
__global__ void k_initbias(float* __restrict__ out, const float* __restrict__ b,
                           int Nn) {
    long i = (long)blockIdx.x * blockDim.x + threadIdx.x;
    if (i >= (long)Nn * 64) return;
    out[i] = b[(int)(i & 63)];
}

// out[d,f] += h[s,f] * (e / (denom[d] + 1e-16))     (F = 64)
// one thread per (edge, 4 features): float4 gather, 4 atomics
__global__ void k_scatter(const float* __restrict__ h, const float* __restrict__ ealpha,
                          const float* __restrict__ denom,
                          const int* __restrict__ src, const int* __restrict__ dst,
                          float* __restrict__ out, int E, int Nn) {
    long i = (long)blockIdx.x * blockDim.x + threadIdx.x;
    long T = (long)(E + Nn) * 16;
    if (i >= T) return;
    int edge = (int)(i >> 4);
    int f    = (int)(i & 15) * 4;
    int s = (edge < E) ? src[edge] : (edge - E);
    int d = (edge < E) ? dst[edge] : (edge - E);
    float coef = ealpha[edge] / (denom[d] + 1e-16f);
    float4 hv = *(const float4*)(h + (size_t)s * 64 + f);
    float* op = out + (size_t)d * 64 + f;
    atomicAdd(op + 0, hv.x * coef);
    atomicAdd(op + 1, hv.y * coef);
    atomicAdd(op + 2, hv.z * coef);
    atomicAdd(op + 3, hv.w * coef);
}

// pooling: sums[batch[n], f] += z[n, f]; counts[batch[n]] += 1 (once per node)
__global__ void k_poolsum(const float* __restrict__ z, const int* __restrict__ batch,
                          float* __restrict__ sums, float* __restrict__ counts, int Nn) {
    long i = (long)blockIdx.x * blockDim.x + threadIdx.x;
    if (i >= (long)Nn * 64) return;
    int n = (int)(i >> 6);
    int f = (int)(i & 63);
    int g = batch[n];
    atomicAdd(&sums[(size_t)g * 64 + f], z[i]);
    if (f == 0) atomicAdd(&counts[g], 1.0f);
}

__global__ void k_pooldiv(const float* __restrict__ sums, const float* __restrict__ counts,
                          float* __restrict__ ge, int G) {
    int i = blockIdx.x * blockDim.x + threadIdx.x;
    if (i >= G * 64) return;
    int g = i >> 6;
    float c = counts[g];
    ge[i] = sums[i] / fmaxf(c, 1.0f);
}

// ---------------------------------------------------------------------------
// Host-side orchestration
// ---------------------------------------------------------------------------
static inline int cdiv(long a, long b) { return (int)((a + b - 1) / b); }

extern "C" void kernel_launch(void* const* d_in, const int* in_sizes, int n_in,
                              void* d_out, int out_size, void* d_ws, size_t ws_size,
                              hipStream_t stream) {
    const float* x       = (const float*)d_in[0];
    const int*   ei      = (const int*)  d_in[1];
    const int*   batch   = (const int*)  d_in[2];
    // d_in[3] = edge_attr (unused: edge_dim=None)
    const float* W1      = (const float*)d_in[4];
    const float* a_src1  = (const float*)d_in[5];
    const float* a_dst1  = (const float*)d_in[6];
    const float* b1      = (const float*)d_in[7];
    const float* W2      = (const float*)d_in[8];
    const float* a_src2  = (const float*)d_in[9];
    const float* a_dst2  = (const float*)d_in[10];
    const float* b2      = (const float*)d_in[11];
    const float* lin_W   = (const float*)d_in[12];
    const float* lin_b   = (const float*)d_in[13];
    const float* xm_W1   = (const float*)d_in[14];
    const float* xm_b1   = (const float*)d_in[15];
    const float* xm_W2   = (const float*)d_in[16];
    const float* xm_b2   = (const float*)d_in[17];
    const float* em_W1   = (const float*)d_in[18];
    const float* em_b1   = (const float*)d_in[19];
    const float* em_W2   = (const float*)d_in[20];
    const float* em_b2   = (const float*)d_in[21];

    const int Nn = in_sizes[0] / 128;   // 50000
    const int E  = in_sizes[1] / 2;     // 800000
    const int G  = 500;                 // N_GRAPHS
    const int T  = E + Nn;              // edges incl. self-loops

    const int* src = ei;
    const int* dst = ei + E;

    // Output layout (tuple order): x_hat[Nn*128], e_hat[Nn*3], z[Nn*64], ge[G*64]
    float* out_xhat = (float*)d_out;
    float* out_ehat = out_xhat + (size_t)Nn * 128;
    float* out_z    = out_ehat + (size_t)Nn * 3;
    float* out_ge   = out_z    + (size_t)Nn * 64;

    // Workspace layout
    float* ws = (float*)d_ws;
    size_t o = 0;
    float* bufA  = ws + o; o += (size_t)Nn * 64;   // linear-transform buffer
    float* bufB  = ws + o; o += (size_t)Nn * 64;   // aggregation buffer
    float* alpha = ws + o; o += (size_t)T;
    float* ssrc  = ws + o; o += Nn;
    float* sdst  = ws + o; o += Nn;
    float* amax  = ws + o; o += Nn;
    float* denom = ws + o; o += Nn;
    float* sums  = ws + o; o += (size_t)G * 64;
    float* cnts  = ws + o; o += G;
    (void)ws_size; (void)n_in; (void)out_size;

    const dim3 blk(256);
    const dim3 gblk(128);                 // 4 waves per block
    const int gM = cdiv(Nn, 64);          // 4 x 16-row tiles per block

    // ---------------- GAT layer 1: h1 = x @ W1 ----------------
    wmma_gemm_kernel<4, 128, false><<<gM, gblk, 0, stream>>>(x, W1, nullptr, bufA, Nn, 64);
    k_scores<<<cdiv(Nn, 256), blk, 0, stream>>>(bufA, a_src1, a_dst1, ssrc, sdst, Nn);
    k_fill<<<cdiv(Nn, 256), blk, 0, stream>>>(amax, -INFINITY, Nn);
    k_fill<<<cdiv(Nn, 256), blk, 0, stream>>>(denom, 0.0f, Nn);
    k_initbias<<<cdiv((long)Nn * 64, 256), blk, 0, stream>>>(bufB, b1, Nn);
    k_alpha<<<cdiv(T, 256), blk, 0, stream>>>(ssrc, sdst, src, dst, alpha, E, Nn);
    k_segmax<<<cdiv(T, 256), blk, 0, stream>>>(alpha, dst, amax, E, Nn);
    k_expsum<<<cdiv(T, 256), blk, 0, stream>>>(alpha, amax, dst, denom, E, Nn);
    k_scatter<<<cdiv((long)T * 16, 256), blk, 0, stream>>>(bufA, alpha, denom, src, dst, bufB, E, Nn);
    k_relu<<<cdiv((long)Nn * 64, 256), blk, 0, stream>>>(bufB, (long)Nn * 64);

    // ---------------- GAT layer 2: h2 = relu(out1) @ W2 ----------------
    wmma_gemm_kernel<4, 64, false><<<gM, gblk, 0, stream>>>(bufB, W2, nullptr, bufA, Nn, 64);
    k_scores<<<cdiv(Nn, 256), blk, 0, stream>>>(bufA, a_src2, a_dst2, ssrc, sdst, Nn);
    k_fill<<<cdiv(Nn, 256), blk, 0, stream>>>(amax, -INFINITY, Nn);
    k_fill<<<cdiv(Nn, 256), blk, 0, stream>>>(denom, 0.0f, Nn);
    k_initbias<<<cdiv((long)Nn * 64, 256), blk, 0, stream>>>(bufB, b2, Nn);
    k_alpha<<<cdiv(T, 256), blk, 0, stream>>>(ssrc, sdst, src, dst, alpha, E, Nn);
    k_segmax<<<cdiv(T, 256), blk, 0, stream>>>(alpha, dst, amax, E, Nn);
    k_expsum<<<cdiv(T, 256), blk, 0, stream>>>(alpha, amax, dst, denom, E, Nn);
    k_scatter<<<cdiv((long)T * 16, 256), blk, 0, stream>>>(bufA, alpha, denom, src, dst, bufB, E, Nn);

    // ---------------- z = out2 @ lin_W + lin_b ----------------
    wmma_gemm_kernel<4, 64, false><<<gM, gblk, 0, stream>>>(bufB, lin_W, lin_b, out_z, Nn, 64);

    // ---------------- global mean pool ----------------
    k_fill<<<cdiv((long)G * 64 + G, 256), blk, 0, stream>>>(sums, 0.0f, (long)G * 64 + G);
    k_poolsum<<<cdiv((long)Nn * 64, 256), blk, 0, stream>>>(out_z, batch, sums, cnts, Nn);
    k_pooldiv<<<cdiv(G * 64, 256), blk, 0, stream>>>(sums, cnts, out_ge, G);

    // ---------------- x decoder: relu(z@xm_W1+b) @ xm_W2 + b ----------------
    wmma_gemm_kernel<4, 64, true ><<<gM, gblk, 0, stream>>>(out_z, xm_W1, xm_b1, bufA, Nn, 64);
    wmma_gemm_kernel<8, 64, false><<<gM, gblk, 0, stream>>>(bufA, xm_W2, xm_b2, out_xhat, Nn, 128);

    // ---------------- edge decoder: relu(z@em_W1+b) @ em_W2 + b ----------------
    wmma_gemm_kernel<4, 64, true ><<<gM, gblk, 0, stream>>>(out_z, em_W1, em_b1, bufA, Nn, 64);
    wmma_gemm_kernel<1, 64, false><<<gM, gblk, 0, stream>>>(bufA, em_W2, em_b2, out_ehat, Nn, 3);
}
